// CausalSelfAttention_88356067213931
// MI455X (gfx1250) — matrix operand order repaired and checked by reading, not compiled, not run on hardware
//
#include <hip/hip_runtime.h>

#define B_  4
#define S_  2048
#define HID 1024
#define NH  16
#define NKV 4
#define HD  64
#define KVW 512   // 2*NKV*HD
#define KD  1024  // contraction depth of both projection GEMMs

#if defined(__has_builtin)
#if __has_builtin(__builtin_amdgcn_global_load_async_to_lds_b128)
#define HAVE_ASYNC_LDS 1
#endif
#endif
#ifndef HAVE_ASYNC_LDS
#define HAVE_ASYNC_LDS 0
#endif

typedef __attribute__((ext_vector_type(16))) __bf16 v16bf;
typedef __attribute__((ext_vector_type(8)))  __bf16 v8bf;
typedef __attribute__((ext_vector_type(8)))  float  v8f;
typedef int v4i __attribute__((vector_size(16)));  // matches builtin param type

static __device__ inline v8f wmma_bf16(v16bf a, v16bf b, v8f c) {
  // D = A(16x32) * B(32x16) + C, f32 accumulate
  return __builtin_amdgcn_wmma_f32_16x16x32_bf16(false, a, false, b, (short)0, c,
                                                 false, false);
}

// A-matrix 16x32 bf16 fragment: lane holds row (lane%16); element j maps to
// k = 8*(lane/16) + 16*(j/8) + (j%8). p = row pointer at k-slice start.
static __device__ inline v16bf frag_a(const __bf16* p, int hi) {
  union { v16bf v; v8bf h[2]; } u;
  u.h[0] = *(const v8bf*)(p + hi * 8);
  u.h[1] = *(const v8bf*)(p + hi * 8 + 16);
  return u.v;
}

// B-matrix 32x16 bf16 fragment: lane holds column n = lane%16 (= row n of B^T);
// element j maps to k = 8*(lane/16) + 16*(j/8) + (j%8). p = B^T row pointer at k-slice start.
static __device__ inline v16bf frag_b(const __bf16* p, int hi) {
  union { v16bf v; v8bf h[2]; } u;
  u.h[0] = *(const v8bf*)(p + hi * 8);
  u.h[1] = *(const v8bf*)(p + hi * 8 + 16);
  return u.v;
}

static __device__ inline v8f vzero() {
  v8f z;
#pragma unroll
  for (int i = 0; i < 8; ++i) z[i] = 0.f;
  return z;
}

// 16B per-lane copy global -> LDS (async path uses ASYNCcnt-tracked DMA)
static __device__ inline void cp16_g2l(const __bf16* g, __bf16* l) {
#if HAVE_ASYNC_LDS
  __builtin_amdgcn_global_load_async_to_lds_b128(
      (__attribute__((address_space(1))) v4i*)(uintptr_t)g,   // inttoptr: global
      (__attribute__((address_space(3))) v4i*)(void*)l,       // addrspacecast
      0, 0);
#else
  *(v8bf*)l = *(const v8bf*)g;
#endif
}

// ---------------------------------------------------------------- converts
__global__ void f32_to_bf16_kernel(const float* __restrict__ src,
                                   __bf16* __restrict__ dst, int n) {
  int i = blockIdx.x * blockDim.x + threadIdx.x;
  if (i < n) dst[i] = (__bf16)src[i];
}

// src [K][N] f32 row-major -> dst [N][K] bf16 row-major (B^T for WMMA)
__global__ void transpose_bf16_kernel(const float* __restrict__ src,
                                      __bf16* __restrict__ dst, int K, int N) {
  int i = blockIdx.x * blockDim.x + threadIdx.x;
  if (i < K * N) {
    int n = i / K, k = i - n * K;
    dst[i] = (__bf16)src[k * N + n];
  }
}

// KVb [b*S][512] (V at col 256 + h*64 + d) -> Vt [b][h][d][s]
__global__ void repack_vt_kernel(const __bf16* __restrict__ kv,
                                 __bf16* __restrict__ vt) {
  int i = blockIdx.x * blockDim.x + threadIdx.x;  // B*NKV*HD*S
  int s = i & (S_ - 1);
  int d = (i >> 11) & (HD - 1);
  int h = (i >> 17) & (NKV - 1);
  int b = i >> 19;
  vt[i] = kv[(size_t)(b * S_ + s) * KVW + NKV * HD + h * HD + d];
}

// ---------------------------------------------------------------- GEMM
// C[M][N] = A[M][KD] * B[KD][N], BT is B^T bf16 [N][KD]. One wave -> 16x64
// tile. Ping-pong pipelined k-loop (2 x 32 per iteration, two stable register
// sets): loads issue a full k-step ahead of their consuming WMMA with no
// register-rotation copies.
__global__ __launch_bounds__(256) void gemm_bf16_kernel(
    const __bf16* __restrict__ A, const __bf16* __restrict__ BT,
    float* __restrict__ Cf, __bf16* __restrict__ Cb, int N) {
  int lane = threadIdx.x & 31, w = threadIdx.x >> 5;
  int lo = lane & 15, hi = lane >> 4;
  int gw = blockIdx.x * 8 + w;
  int tilesN = N >> 6;
  int wm = gw / tilesN, wn = gw - wm * tilesN;

  const __bf16* arow = A + (size_t)(wm * 16 + lo) * KD;
  const __bf16* brow = BT + (size_t)(wn * 64 + lo) * KD;  // +t*16*KD per tile
  v8f acc[4];
#pragma unroll
  for (int t = 0; t < 4; ++t) acc[t] = vzero();

  // set A preloaded at k=0
  v16bf afA = frag_a(arow, hi);
  v16bf bfA[4];
#pragma unroll
  for (int t = 0; t < 4; ++t) bfA[t] = frag_b(brow + (size_t)t * 16 * KD, hi);

  for (int k = 0; k < KD; k += 64) {
    // set B: fetch k+32 (fresh defs -> loads land in the consuming registers)
    v16bf afB = frag_a(arow + k + 32, hi);
    v16bf bfB[4];
#pragma unroll
    for (int t = 0; t < 4; ++t)
      bfB[t] = frag_b(brow + (size_t)t * 16 * KD + k + 32, hi);

#pragma unroll
    for (int t = 0; t < 4; ++t) acc[t] = wmma_bf16(afA, bfA[t], acc[t]);

    if (k + 64 < KD) {  // uniform scalar branch; reload set A in place
      afA = frag_a(arow + k + 64, hi);
#pragma unroll
      for (int t = 0; t < 4; ++t)
        bfA[t] = frag_b(brow + (size_t)t * 16 * KD + k + 64, hi);
    }
#pragma unroll
    for (int t = 0; t < 4; ++t) acc[t] = wmma_bf16(afB, bfB[t], acc[t]);
  }

  // C layout: VGPR r -> row wm*16 + r + 8*hi, col = wn*64 + t*16 + lane%16
#pragma unroll
  for (int t = 0; t < 4; ++t)
#pragma unroll
    for (int r = 0; r < 8; ++r) {
      size_t idx = (size_t)(wm * 16 + r + 8 * hi) * N + wn * 64 + t * 16 + lo;
      if (Cf) Cf[idx] = acc[t][r];
      else    Cb[idx] = (__bf16)acc[t][r];
    }
}

// ---------------------------------------------------------------- attention
// One block (8 waves) = one (b, head, 128-query slab); each wave owns 16
// queries. K/V chunks (32 keys) staged into double-buffered LDS once per
// block (8x L2 traffic reduction) via async global->LDS when available.
__global__ __launch_bounds__(256) void attn_kernel(
    const __bf16* __restrict__ Qb, const __bf16* __restrict__ KVb,
    const __bf16* __restrict__ Vt, __bf16* __restrict__ Ab) {
  __shared__ __attribute__((aligned(16))) __bf16 Kbuf[2][32][64];  // [key][dim]
  __shared__ __attribute__((aligned(16))) __bf16 Vbuf[2][64][32];  // [dim][key]
  __shared__ __attribute__((aligned(16))) __bf16 Pt[8][16][32];    // per-wave P

  int tid = threadIdx.x;
  int lane = tid & 31, w = tid >> 5;
  int lo = lane & 15, hi = lane >> 4;

  int qb = blockIdx.x & 15;            // S/128 slabs
  int head = (blockIdx.x >> 4) & 15;
  int b = blockIdx.x >> 8;
  int hk = head >> 2;                  // kv head (g = 4)
  int qbase = (qb * 8 + w) * 16;
  float slope = exp2f(-0.5f * (float)(head + 1));  // 256^(1/16) = 2^0.5

  // Q A-fragments (dims 0..31, 32..63) hoisted out of the KV loop
  const __bf16* qrow = Qb + (size_t)(b * S_ + qbase + lo) * HID + head * HD;
  v16bf aq0 = frag_a(qrow, hi);
  v16bf aq1 = frag_a(qrow + 32, hi);

  const __bf16* kbase = KVb + (size_t)b * S_ * KVW + hk * HD;   // + key*KVW
  const __bf16* vbase = Vt + (size_t)(b * NKV + hk) * HD * S_;  // + d*S + key

  // per-thread staging slots: 256 threads x (16B of K + 16B of V) per chunk
  int kkey = tid >> 3, kseg = tid & 7;  // K chunk: 32 rows x 128B
  int vdim = tid >> 2, vseg = tid & 3;  // V chunk: 64 rows x 64B
  auto stage = [&](int buf, int kv) {
    cp16_g2l(kbase + (size_t)(kv + kkey) * KVW + kseg * 8,
             &Kbuf[buf][kkey][kseg * 8]);
    cp16_g2l(vbase + (size_t)vdim * S_ + kv + vseg * 8,
             &Vbuf[buf][vdim][vseg * 8]);
  };

  float rowmax[8], rowsum[8];
#pragma unroll
  for (int r = 0; r < 8; ++r) { rowmax[r] = -1e30f; rowsum[r] = 0.f; }
  v8f oacc[4];
#pragma unroll
  for (int t = 0; t < 4; ++t) oacc[t] = vzero();

  stage(0, 0);  // prologue: chunk 0 -> buffer 0

  for (int kv = 0, it = 0; kv < S_; kv += 32, ++it) {
    int cur = it & 1;
    __syncthreads();  // prior compute done reading buf cur^1 -> safe to refill
    bool more = (kv + 32) < S_;
    if (more) stage(cur ^ 1, kv + 32);
#if HAVE_ASYNC_LDS
    if (more) asm volatile("s_wait_asynccnt 0x2" ::: "memory");
    else      asm volatile("s_wait_asynccnt 0x0" ::: "memory");
#endif
    __syncthreads();  // all waves' stage of chunk `cur` visible

    // scores S = Q * K^T : two 16-key groups, contraction 64 dims in 2 steps
    const __bf16* kr0 = &Kbuf[cur][lo][0];
    const __bf16* kr1 = &Kbuf[cur][16 + lo][0];
    v8f s0 = vzero(), s1 = vzero();
    s0 = wmma_bf16(aq0, frag_b(kr0, hi), s0);
    s0 = wmma_bf16(aq1, frag_b(kr0 + 32, hi), s0);
    s1 = wmma_bf16(aq0, frag_b(kr1, hi), s1);
    s1 = wmma_bf16(aq1, frag_b(kr1 + 32, hi), s1);

    // online softmax; row m = qbase + r + 8*hi lives across 16 lanes of a half
#pragma unroll
    for (int r = 0; r < 8; ++r) {
      float m = (float)(qbase + r + 8 * hi);
      float f0 = s0[r] * 0.125f + slope * ((float)(kv + lo) - m);
      float f1 = s1[r] * 0.125f + slope * ((float)(kv + 16 + lo) - m);
      float cmax = fmaxf(f0, f1);
      for (int d = 1; d < 16; d <<= 1) cmax = fmaxf(cmax, __shfl_xor(cmax, d, 32));
      float nm = fmaxf(rowmax[r], cmax);
      float corr = __expf(rowmax[r] - nm);
      rowmax[r] = nm;
      float p0 = __expf(f0 - nm), p1 = __expf(f1 - nm);
      float ps = p0 + p1;
      for (int d = 1; d < 16; d <<= 1) ps += __shfl_xor(ps, d, 32);
      rowsum[r] = rowsum[r] * corr + ps;
#pragma unroll
      for (int t = 0; t < 4; ++t) oacc[t][r] *= corr;
      // C-layout -> [q][k] tile in LDS (bf16) for the P A-fragment
      Pt[w][r + 8 * hi][lo]      = (__bf16)p0;
      Pt[w][r + 8 * hi][16 + lo] = (__bf16)p1;
    }
    asm volatile("s_wait_dscnt 0x0" ::: "memory");  // same-wave DS in-order
    v16bf pf = frag_a(&Pt[w][lo][0], hi);           // P: 16q x 32keys

    // O += P * V : contraction over 32 keys, 4 dim tiles of 16
#pragma unroll
    for (int t = 0; t < 4; ++t) {
      oacc[t] = wmma_bf16(pf, frag_b(&Vbuf[cur][t * 16 + lo][0], hi), oacc[t]);
    }
  }

#pragma unroll
  for (int t = 0; t < 4; ++t)
#pragma unroll
    for (int r = 0; r < 8; ++r) {
      size_t idx = (size_t)(b * S_ + qbase + r + 8 * hi) * HID + head * HD +
                   t * 16 + lo;
      Ab[idx] = (__bf16)(oacc[t][r] / rowsum[r]);
    }
}

// ---------------------------------------------------------------- launch
extern "C" void kernel_launch(void* const* d_in, const int* in_sizes, int n_in,
                              void* d_out, int out_size, void* d_ws,
                              size_t ws_size, hipStream_t stream) {
  const float* x   = (const float*)d_in[0];
  const float* Wq  = (const float*)d_in[1];
  const float* Wkv = (const float*)d_in[2];
  const float* Wo  = (const float*)d_in[3];
  float* y = (float*)d_out;

  char* ws = (char*)d_ws;
  size_t off = 0;
  auto wsalloc = [&](size_t bytes) -> void* {
    void* p = ws + off;
    off += (bytes + 255) & ~(size_t)255;
    return p;
  };
  __bf16* xb   = (__bf16*)wsalloc((size_t)B_ * S_ * HID * 2);
  __bf16* WqT  = (__bf16*)wsalloc((size_t)HID * HID * 2);
  __bf16* WkvT = (__bf16*)wsalloc((size_t)HID * KVW * 2);
  __bf16* WoT  = (__bf16*)wsalloc((size_t)HID * HID * 2);
  __bf16* Qb   = (__bf16*)wsalloc((size_t)B_ * S_ * HID * 2);
  __bf16* KVb  = (__bf16*)wsalloc((size_t)B_ * S_ * KVW * 2);
  __bf16* Vtb  = (__bf16*)wsalloc((size_t)B_ * NKV * HD * S_ * 2);
  __bf16* Ab   = (__bf16*)wsalloc((size_t)B_ * S_ * HID * 2);

  const int M = B_ * S_;
  const int nx = M * HID;
  f32_to_bf16_kernel<<<(nx + 255) / 256, 256, 0, stream>>>(x, xb, nx);
  transpose_bf16_kernel<<<(HID * HID + 255) / 256, 256, 0, stream>>>(Wq, WqT, HID, HID);
  transpose_bf16_kernel<<<(HID * KVW + 255) / 256, 256, 0, stream>>>(Wkv, WkvT, HID, KVW);
  transpose_bf16_kernel<<<(HID * HID + 255) / 256, 256, 0, stream>>>(Wo, WoT, HID, HID);

  // Q = x @ Wq  (bf16 out)
  gemm_bf16_kernel<<<(M / 16) * (HID / 64) / 8, 256, 0, stream>>>(
      xb, WqT, (float*)nullptr, Qb, HID);
  // KV = x @ Wkv (bf16 out)
  gemm_bf16_kernel<<<(M / 16) * (KVW / 64) / 8, 256, 0, stream>>>(
      xb, WkvT, (float*)nullptr, KVb, KVW);
  // V transpose repack
  repack_vt_kernel<<<(B_ * NKV * HD * S_) / 256, 256, 0, stream>>>(KVb, Vtb);
  // attention: one block per (b, head, 128 queries)
  attn_kernel<<<B_ * NH * (S_ / 128), 256, 0, stream>>>(Qb, KVb, Vtb, Ab);
  // y = attn @ Wo (f32 out)
  gemm_bf16_kernel<<<(M / 16) * (HID / 64) / 8, 256, 0, stream>>>(
      Ab, WoT, y, (__bf16*)nullptr, HID);

  (void)in_sizes; (void)n_in; (void)out_size; (void)ws_size;
}
